// SCAM_48971217109020
// MI455X (gfx1250) — matrix operand order repaired and checked by reading, not compiled, not run on hardware
//
#include <hip/hip_runtime.h>
#include <hip/hip_bf16.h>
#include <math.h>

// Problem constants (from reference): H=W=48, WS=8 -> N=2304, K=289, D=256, 8 heads x 32
#define N_TOK   2304
#define D_MODEL 256
#define K_NB    289
#define HD      32

typedef __attribute__((ext_vector_type(16))) __bf16 v16bf;
typedef __attribute__((ext_vector_type(8)))  float  v8f;

union FragBF { v16bf v; __bf16 e[16]; unsigned u[8]; };
union FragF  { v8f   v; float  f[8]; };

// A-fragment (16x32 bf16) K offset for VGPR v, lane-half hf (contiguous order):
// lanes 0-15: V0..7 -> K=0..15 pairs ; lanes 16-31: K=16..31
__device__ __forceinline__ int a_koff(int v, int hf) {
    return hf * 16 + (v << 1);
}

// ---------------------------------------------------------------- converts
__global__ void cvt_f32_to_bf16(const float* __restrict__ in,
                                __bf16* __restrict__ out, int n) {
    int i = blockIdx.x * blockDim.x + threadIdx.x;
    if (i < n) out[i] = (__bf16)in[i];
}

// ------------------------------------------- neighborhood attn + layernorm
__global__ void neigh_fuse_ln(const float* __restrict__ x2,
                              const int*   __restrict__ nidx,
                              const float* __restrict__ gamma,
                              const float* __restrict__ beta,
                              __bf16* __restrict__ x2f_bf) {
    __shared__ float sx[D_MODEL];
    __shared__ float s[K_NB];
    __shared__ int   nbs[K_NB];
    __shared__ float red[256];
    const int n = blockIdx.x;
    const int t = threadIdx.x;

    sx[t] = x2[(long)n * D_MODEL + t];
    for (int k = t; k < K_NB; k += 256) nbs[k] = nidx[n * K_NB + k];
    __syncthreads();

    // scores[k] = dot(x2[n], x2[nb_k]) / 16 ; 8 waves, one neighbor per wave
    const int wave = t >> 5, lane = t & 31;
    for (int k = wave; k < K_NB; k += 8) {
        const float* xr = x2 + (long)nbs[k] * D_MODEL;
        float p = 0.f;
        #pragma unroll
        for (int i = 0; i < 8; ++i) p += sx[lane + 32 * i] * xr[lane + 32 * i];
        #pragma unroll
        for (int off = 16; off >= 1; off >>= 1) p += __shfl_xor(p, off, 32);
        if (lane == 0) s[k] = p * 0.0625f;   // 1/sqrt(256)
    }
    __syncthreads();

    // softmax over K_NB
    float lm = -__builtin_inff();
    for (int k = t; k < K_NB; k += 256) lm = fmaxf(lm, s[k]);
    red[t] = lm; __syncthreads();
    for (int st = 128; st >= 1; st >>= 1) { if (t < st) red[t] = fmaxf(red[t], red[t + st]); __syncthreads(); }
    const float mx = red[0]; __syncthreads();
    float ls = 0.f;
    for (int k = t; k < K_NB; k += 256) { float e = __expf(s[k] - mx); s[k] = e; ls += e; }
    red[t] = ls; __syncthreads();
    for (int st = 128; st >= 1; st >>= 1) { if (t < st) red[t] += red[t + st]; __syncthreads(); }
    const float inv = 1.f / red[0]; __syncthreads();

    // weighted accumulate, one feature dim per thread
    float acc = 0.f;
    for (int k = 0; k < K_NB; ++k) {
        const float* xr = x2 + (long)nbs[k] * D_MODEL;
        if (k + 1 < K_NB)
            __builtin_prefetch(&x2[(long)nbs[k + 1] * D_MODEL + t], 0, 0);
        acc += s[k] * xr[t];
    }
    acc *= inv;

    // layernorm over D (population variance, eps 1e-5)
    red[t] = acc; __syncthreads();
    for (int st = 128; st >= 1; st >>= 1) { if (t < st) red[t] += red[t + st]; __syncthreads(); }
    const float mu = red[0] * (1.f / 256.f); __syncthreads();
    const float dv = acc - mu;
    red[t] = dv * dv; __syncthreads();
    for (int st = 128; st >= 1; st >>= 1) { if (t < st) red[t] += red[t + st]; __syncthreads(); }
    const float var = red[0] * (1.f / 256.f); __syncthreads();
    const float y = dv * rsqrtf(var + 1e-5f) * gamma[t] + beta[t];
    x2f_bf[(long)n * D_MODEL + t] = (__bf16)y;
}

// ------------------------------------------------------- QKV projection GEMM
// qkv[n][c] = (c<256 ? x2f : x1)[n][:] . w[c][:] + b[c] ; one wave per 16x16 tile.
// V columns additionally stored transposed (vt[channel][n]) so the attention
// kernel can load V B-fragments as contiguous dwords.
__global__ void qkv_gemm(const __bf16* __restrict__ x2f_bf,
                         const __bf16* __restrict__ x1_bf,
                         const __bf16* __restrict__ w_bf,   // [768][256]
                         const float*  __restrict__ bias,   // [768]
                         __bf16* __restrict__ qkv_bf,       // [N][768]
                         __bf16* __restrict__ vt_bf) {      // [256][N]
    const int tm = blockIdx.x;        // 0..143
    const int tn = blockIdx.y;        // 0..47
    const int lane = threadIdx.x;
    const int hf = lane >> 4, l16 = lane & 15;
    const __bf16* A = (tn < 16) ? x2f_bf : x1_bf;

    FragF acc;
    #pragma unroll
    for (int r = 0; r < 8; ++r) acc.f[r] = 0.f;
    const int arow = tm * 16 + l16;
    const int ncol = tn * 16 + l16;

    for (int kb = 0; kb < 256; kb += 32) {
        FragBF a, b;
        #pragma unroll
        for (int v = 0; v < 8; ++v) {
            a.u[v] = *reinterpret_cast<const unsigned*>(&A[(long)arow * 256 + kb + a_koff(v, hf)]);
            b.u[v] = *reinterpret_cast<const unsigned*>(&w_bf[(long)ncol * 256 + kb + hf * 16 + 2 * v]);
        }
        acc.v = __builtin_amdgcn_wmma_f32_16x16x32_bf16(false, a.v, false, b.v,
                                                        (short)0, acc.v, false, false);
    }
    const float bc = bias[ncol];
    const bool is_v = (tn >= 32);
    #pragma unroll
    for (int r = 0; r < 8; ++r) {
        const int row = tm * 16 + hf * 8 + r;
        const __bf16 y = (__bf16)(acc.f[r] + bc);
        qkv_bf[(long)row * 768 + ncol] = y;
        if (is_v)
            vt_bf[(long)(ncol - 512) * N_TOK + row] = y;   // keys contiguous
    }
}

// ------------------------------------------------ flash multi-head attention
// one wave per (16-query tile, head); online softmax over key tiles of 32
__global__ void mha_flash(const __bf16* __restrict__ qkv_bf,  // [N][768]
                          const __bf16* __restrict__ vt_bf,   // [256][N]
                          __bf16* __restrict__ o_bf) {        // [N][256]
    __shared__ __align__(16) __bf16 pls[16 * 32];
    const int qt = blockIdx.x;     // 0..143
    const int h  = blockIdx.y;     // 0..7
    const int lane = threadIdx.x;
    const int hf = lane >> 4, l16 = lane & 15;
    const float scale = 0.17677669529663687f;   // 1/sqrt(32)

    // Q fragment (A: 16x32 bf16), rows qt*16..+15, cols h*32..+31
    FragBF qa;
    {
        const long base = (long)(qt * 16 + l16) * 768 + h * 32;
        #pragma unroll
        for (int v = 0; v < 8; ++v)
            qa.u[v] = *reinterpret_cast<const unsigned*>(&qkv_bf[base + a_koff(v, hf)]);
    }
    FragF o0, o1;
    float mrow[8], lrow[8];
    #pragma unroll
    for (int r = 0; r < 8; ++r) {
        o0.f[r] = 0.f; o1.f[r] = 0.f;
        mrow[r] = -__builtin_inff(); lrow[r] = 0.f;
    }

    // transposed-V row bases for this lane's B-fragment columns
    const __bf16* vt0 = vt_bf + (long)(h * 32 + l16) * N_TOK;        // hd col  l16
    const __bf16* vt1 = vt_bf + (long)(h * 32 + 16 + l16) * N_TOK;   // hd col  16+l16

    for (int kt = 0; kt < N_TOK; kt += 32) {
        // K^T fragments (B: 32x16): column = key, K-dim = head dim (contiguous pairs)
        FragBF bk0, bk1;
        #pragma unroll
        for (int v = 0; v < 8; ++v) {
            const int d = hf * 16 + 2 * v;
            bk0.u[v] = *reinterpret_cast<const unsigned*>(&qkv_bf[(long)(kt      + l16) * 768 + 256 + h * 32 + d]);
            bk1.u[v] = *reinterpret_cast<const unsigned*>(&qkv_bf[(long)(kt + 16 + l16) * 768 + 256 + h * 32 + d]);
        }
        v8f z = {};
        FragF s0, s1;
        s0.v = __builtin_amdgcn_wmma_f32_16x16x32_bf16(false, qa.v, false, bk0.v, (short)0, z, false, false);
        s1.v = __builtin_amdgcn_wmma_f32_16x16x32_bf16(false, qa.v, false, bk1.v, (short)0, z, false, false);

        // online softmax per query row (rows live on 16-lane halves)
        #pragma unroll
        for (int r = 0; r < 8; ++r) {
            const float v0 = s0.f[r] * scale;
            const float v1 = s1.f[r] * scale;
            float rmax = fmaxf(v0, v1);
            #pragma unroll
            for (int off = 8; off >= 1; off >>= 1) rmax = fmaxf(rmax, __shfl_xor(rmax, off, 32));
            const float mnew = fmaxf(mrow[r], rmax);
            const float corr = __expf(mrow[r] - mnew);
            const float p0 = __expf(v0 - mnew);
            const float p1 = __expf(v1 - mnew);
            float rs = p0 + p1;
            #pragma unroll
            for (int off = 8; off >= 1; off >>= 1) rs += __shfl_xor(rs, off, 32);
            lrow[r] = lrow[r] * corr + rs;
            mrow[r] = mnew;
            o0.f[r] *= corr;
            o1.f[r] *= corr;
            const int row = hf * 8 + r;
            pls[row * 32 + l16]      = (__bf16)p0;
            pls[row * 32 + 16 + l16] = (__bf16)p1;
        }
        __syncthreads();

        // P as A-fragment (16 q-rows x 32 keys) from LDS
        FragBF pa, bv0, bv1;
        #pragma unroll
        for (int v = 0; v < 8; ++v)
            pa.u[v] = *reinterpret_cast<const unsigned*>(&pls[l16 * 32 + a_koff(v, hf)]);
        // V fragments (B: 32 keys x 16 hd-cols) from transposed V: contiguous pairs
        #pragma unroll
        for (int v = 0; v < 8; ++v) {
            const int key = kt + hf * 16 + 2 * v;
            bv0.u[v] = *reinterpret_cast<const unsigned*>(&vt0[key]);
            bv1.u[v] = *reinterpret_cast<const unsigned*>(&vt1[key]);
        }
        o0.v = __builtin_amdgcn_wmma_f32_16x16x32_bf16(false, pa.v, false, bv0.v, (short)0, o0.v, false, false);
        o1.v = __builtin_amdgcn_wmma_f32_16x16x32_bf16(false, pa.v, false, bv1.v, (short)0, o1.v, false, false);
        __syncthreads();
    }

    #pragma unroll
    for (int r = 0; r < 8; ++r) {
        const float invl = 1.f / lrow[r];
        const long row = (long)(qt * 16 + hf * 8 + r);
        o_bf[row * 256 + h * 32 + l16]      = (__bf16)(o0.f[r] * invl);
        o_bf[row * 256 + h * 32 + 16 + l16] = (__bf16)(o1.f[r] * invl);
    }
}

// ------------------------------------------------------- output projection
__global__ void out_gemm(const __bf16* __restrict__ A,      // [N][256]
                         const __bf16* __restrict__ W,      // [256][256]
                         const float*  __restrict__ bias,   // [256]
                         float* __restrict__ out) {         // [N][256] f32
    const int tm = blockIdx.x;     // 0..143
    const int tn = blockIdx.y;     // 0..15
    const int lane = threadIdx.x;
    const int hf = lane >> 4, l16 = lane & 15;

    FragF acc;
    #pragma unroll
    for (int r = 0; r < 8; ++r) acc.f[r] = 0.f;
    const int arow = tm * 16 + l16;
    const int ncol = tn * 16 + l16;

    for (int kb = 0; kb < 256; kb += 32) {
        FragBF a, b;
        #pragma unroll
        for (int v = 0; v < 8; ++v) {
            a.u[v] = *reinterpret_cast<const unsigned*>(&A[(long)arow * 256 + kb + a_koff(v, hf)]);
            b.u[v] = *reinterpret_cast<const unsigned*>(&W[(long)ncol * 256 + kb + hf * 16 + 2 * v]);
        }
        acc.v = __builtin_amdgcn_wmma_f32_16x16x32_bf16(false, a.v, false, b.v,
                                                        (short)0, acc.v, false, false);
    }
    const float bc = bias[ncol];
    #pragma unroll
    for (int r = 0; r < 8; ++r) {
        const int row = tm * 16 + hf * 8 + r;
        out[(long)row * 256 + ncol] = acc.f[r] + bc;
    }
}

// --------------------------------------------------------------------------
extern "C" void kernel_launch(void* const* d_in, const int* in_sizes, int n_in,
                              void* d_out, int out_size, void* d_ws, size_t ws_size,
                              hipStream_t stream) {
    const float* x1    = (const float*)d_in[0];
    const float* x2    = (const float*)d_in[1];
    const float* ln_g  = (const float*)d_in[2];
    const float* ln_b  = (const float*)d_in[3];
    const float* w_in  = (const float*)d_in[4];   // [768][256]
    const float* b_in  = (const float*)d_in[5];   // [768]
    const float* w_out = (const float*)d_in[6];   // [256][256]
    const float* b_out = (const float*)d_in[7];   // [256]
    const int*   nidx  = (const int*)d_in[8];     // [2304][289]
    float* out = (float*)d_out;

    char* ws = (char*)d_ws;
    __bf16* x1_bf   = (__bf16*)(ws + 0);          // 1,179,648 B
    __bf16* x2f_bf  = (__bf16*)(ws + 1179648);    // 1,179,648 B
    __bf16* wqkv_bf = (__bf16*)(ws + 2359296);    //   393,216 B
    __bf16* wout_bf = (__bf16*)(ws + 2752512);    //   131,072 B
    __bf16* qkv_bf  = (__bf16*)(ws + 2883584);    // 3,538,944 B
    __bf16* o_bf    = (__bf16*)(ws + 6422528);    // 1,179,648 B
    __bf16* vt_bf   = (__bf16*)(ws + 7602176);    // 1,179,648 B  (total ~8.8 MB)

    cvt_f32_to_bf16<<<(589824 + 255) / 256, 256, 0, stream>>>(x1,    x1_bf,   589824);
    cvt_f32_to_bf16<<<(196608 + 255) / 256, 256, 0, stream>>>(w_in,  wqkv_bf, 196608);
    cvt_f32_to_bf16<<<( 65536 + 255) / 256, 256, 0, stream>>>(w_out, wout_bf,  65536);

    neigh_fuse_ln<<<N_TOK, 256, 0, stream>>>(x2, nidx, ln_g, ln_b, x2f_bf);

    qkv_gemm<<<dim3(N_TOK / 16, 48), 32, 0, stream>>>(x2f_bf, x1_bf, wqkv_bf, b_in,
                                                      qkv_bf, vt_bf);

    mha_flash<<<dim3(N_TOK / 16, 8), 32, 0, stream>>>(qkv_bf, vt_bf, o_bf);

    out_gemm<<<dim3(N_TOK / 16, 16), 32, 0, stream>>>(o_bf, wout_bf, b_out, out);
}
